// TransMilPlipMoe_80418967650851
// MI455X (gfx1250) — matrix operand order repaired and checked by reading, not compiled, not run on hardware
//
#include <hip/hip_runtime.h>
#include <hip/hip_bf16.h>

// ---------------- problem constants ----------------
constexpr int B = 8, S = 4096, D = 512, F = 2048, E = 8;
constexpr int ST = 32;                 // s-rows per block tile
constexpr int SBLK = S / ST;           // 128 s-tiles per batch sample
constexpr float LN_EPS = 1e-5f;
constexpr float LN_MERGE = 0.03f;

// padded LDS leading dims (keep 16B alignment: pads are multiples of 8 halves / 4 floats)
constexpr int LDX = D + 8;             // x tile, f16   (row = 1024B data + 16B pad)
constexpr int LDH = F + 8;             // h tile, f16
constexpr int LDO = D + 4;             // out tile, f32

typedef _Float16 v16h __attribute__((ext_vector_type(16)));
typedef _Float16 v8h  __attribute__((ext_vector_type(8)));
typedef _Float16 v4h  __attribute__((ext_vector_type(4)));
typedef float    v8f  __attribute__((ext_vector_type(8)));
typedef unsigned int u32x4 __attribute__((ext_vector_type(4)));
typedef int          i32x4 __attribute__((ext_vector_type(4)));
typedef int          i32x8 __attribute__((ext_vector_type(8)));

// ---------------- WMMA fragment helpers ----------------
// A fragment (16x32 f16) from row-major LDS tile.
// Contiguous k order used here (B matches): lanes 0-15 -> M=lane, K=[0..15]; lanes 16-31 -> M=lane-16, K=[16..31]
__device__ __forceinline__ v16h load_a_lds(const _Float16* base, int ld, int m0, int k0, int lane) {
    const int ml = lane & 15;
    const int hs = (lane >> 4) & 1;
    const _Float16* p = base + (m0 + ml) * ld + k0 + hs * 16;
    v8h lo = *(const v8h*)p;
    v8h hi = *(const v8h*)(p + 8);
    return __builtin_shufflevector(lo, hi, 0, 1, 2, 3, 4, 5, 6, 7, 8, 9, 10, 11, 12, 13, 14, 15);
}

__device__ __forceinline__ v8f wmma_f16(v16h a, v16h b, v8f c) {
    return __builtin_amdgcn_wmma_f32_16x16x32_f16(false, a, false, b, (short)0, c, false, false);
}

// ---------------- pipelined k-outer GEMM: 8 C tiles per wave ----------------
// A from LDS (one frag per k-step, shared by 8 WMMAs); B from global, row-major [N][K].
// One-deep software pipeline on both A and B so loads overlap the matrix pipe
// (compiler can issue partial s_wait_loadcnt instead of waiting to zero).
// B row for tile j is (n0base + j*nstep); nstep in matrix rows.
__device__ __forceinline__ void gemm8_pipelined(const _Float16* albase, int ald, int m0,
                                                const _Float16* bbase, size_t bld,
                                                int n0base, int nstep, int ktiles,
                                                int lane, v8f acc[8]) {
    const int nl   = lane & 15;
    const int hsel = (lane >> 4) & 1;
    const _Float16* brow = bbase + (size_t)(n0base + nl) * bld + hsel * 16;
    const size_t    jstr = (size_t)nstep * bld;          // per-j row stride (elements)

    v16h a  = load_a_lds(albase, ald, m0, 0, lane);
    v16h bc = *(const v16h*)brow;
#pragma unroll 2
    for (int kk = 0; kk < ktiles; ++kk) {
        v16h an = (kk + 1 < ktiles) ? load_a_lds(albase, ald, m0, (kk + 1) * 32, lane) : a;
        if (kk + 2 < ktiles)
            __builtin_prefetch((const void*)(brow + (kk + 2) * 32), 0, 0);
#pragma unroll
        for (int j = 0; j < 8; ++j) {
            v16h bn;
            if (j < 7)
                bn = *(const v16h*)(brow + (size_t)(j + 1) * jstr + kk * 32);
            else if (kk + 1 < ktiles)
                bn = *(const v16h*)(brow + (kk + 1) * 32);
            else
                bn = bc;
            acc[j] = wmma_f16(a, bc, acc[j]);
            bc = bn;
        }
        a = an;
    }
}

// ---------------- TDM: stage a [ST x D] f16 tile into LDS with row padding ----------------
// D# per CDNA5 ISA ch.8: group0 = {flags/count, lds_addr, global_addr, type},
// group1 = {mask/data_size/pad, dims, tile dims, strides}. pad_interval=7 (256 DWORDs
// = one 1024B row), pad_amount=3 (4 DWORDs = 16B) reproduces the LDX = D+8 layout.
__device__ __forceinline__ void tdm_load_x_tile(const _Float16* gsrc, unsigned lds_off) {
#if __has_builtin(__builtin_amdgcn_tensor_load_to_lds)
    unsigned long long ga = (unsigned long long)(uintptr_t)gsrc;
    u32x4 g0;
    g0.x = 1u;                                        // count=1, user mode
    g0.y = lds_off;                                   // lds_addr
    g0.z = (unsigned)(ga & 0xffffffffu);              // global_addr[31:0]
    g0.w = (unsigned)((ga >> 32) & 0x1ffffffu) | (2u << 30);  // global_addr[56:32] | type=2
    i32x8 g1;
    g1[0] = (1 << 16)                                 // data_size = 2 bytes
          | (1 << 20)                                 // pad_enable
          | (7 << 22)                                 // pad_interval = 256 DWORDs
          | (3 << 25);                                // pad_amount  = 4 DWORDs
    g1[1] = (int)(D << 16);                           // tensor_dim0[15:0] at bits 63:48
    g1[2] = (int)0x80000000u;                         // tensor_dim1[15:0]=32768 at bits 95:80
    g1[3] = (int)(D << 16);                           // tile_dim0 = 512 at bits 127:112
    g1[4] = ST;                                       // tile_dim1 = 32
    g1[5] = D;                                        // tensor_dim0_stride = 512 elements
    g1[6] = 0;
    g1[7] = 0;
    i32x4 gz = {0, 0, 0, 0};
#if defined(__clang_major__) && (__clang_major__ >= 23)
    i32x8 gz8 = {0, 0, 0, 0, 0, 0, 0, 0};
    __builtin_amdgcn_tensor_load_to_lds(g0, g1, gz, gz, gz8, 0);
#else
    __builtin_amdgcn_tensor_load_to_lds(g0, g1, gz, gz, 0);
#endif
    __builtin_amdgcn_s_wait_tensorcnt(0);
#endif
}

// ---------------- utility kernels ----------------
__global__ void cvt_f32_to_f16_kernel(const float* __restrict__ src,
                                      _Float16* __restrict__ dst, int n4) {
    int i = blockIdx.x * blockDim.x + threadIdx.x;
    if (i < n4) {
        float4 v = ((const float4*)src)[i];
        v4h h;
        h.x = (_Float16)v.x; h.y = (_Float16)v.y;
        h.z = (_Float16)v.z; h.w = (_Float16)v.w;
        ((v4h*)dst)[i] = h;
    }
}

__global__ void zero_f32_kernel(float* __restrict__ p, int n) {
    int i = blockIdx.x * blockDim.x + threadIdx.x;
    if (i < n) p[i] = 0.f;
}

// ---------------- gate: gh = relu(x @ gfc1^T + b); gsum[b,e] += sum_s gh @ gfc2^T ----------------
__global__ void __launch_bounds__(256, 1)
gate_kernel(const _Float16* __restrict__ xh,
            const _Float16* __restrict__ w1g,     // [D(h)][D(d)] f16
            const float* __restrict__ gfc1_b,     // [D]
            const float* __restrict__ gfc2_w,     // [E][D]
            float* __restrict__ gsum) {           // [B][E]
    __shared__ _Float16 xs[ST * LDX];
    __shared__ float    sg[E * D];
    __shared__ float    sred[E];

    const int tid  = threadIdx.x;
    const int lane = tid & 31;
    const int wid  = tid >> 5;
    const int b    = blockIdx.x >> 7;        // /128
    const int st   = blockIdx.x & 127;
    const int s0   = st * ST;
    const int nl   = lane & 15;

    const size_t xbase = ((size_t)b * S + s0) * D;
    for (int c = tid; c < ST * D / 8; c += 256) {
        int row = c >> 6;
        int col = (c & 63) * 8;
        *(v8h*)(xs + row * LDX + col) = *(const v8h*)(xh + xbase + row * D + col);
    }
    for (int i = tid; i < E * D; i += 256) sg[i] = gfc2_w[i];
    if (tid < E) sred[tid] = 0.f;
    __syncthreads();

    // each wave owns 8 C tiles: M strip = parity of wid, N tiles (wid>>1) + 4j
    const int m0 = (wid & 1) * 16;
    v8f acc[8];
#pragma unroll
    for (int j = 0; j < 8; ++j) acc[j] = (v8f){};

    gemm8_pipelined(xs, LDX, m0, w1g, D, (wid >> 1) * 16, 64, D / 32, lane, acc);

    float pacc[E];
#pragma unroll
    for (int e = 0; e < E; ++e) pacc[e] = 0.f;
#pragma unroll
    for (int j = 0; j < 8; ++j) {
        const int n0 = ((wid >> 1) + 4 * j) * 16;
        const float bias = gfc1_b[n0 + nl];
        float rowsum = 0.f;
#pragma unroll
        for (int r = 0; r < 8; ++r) rowsum += fmaxf(acc[j][r] + bias, 0.f);
#pragma unroll
        for (int e = 0; e < E; ++e) pacc[e] += rowsum * sg[e * D + n0 + nl];
    }

#pragma unroll
    for (int e = 0; e < E; ++e) atomicAdd(&sred[e], pacc[e]);
    __syncthreads();
    if (tid < E) atomicAdd(&gsum[b * E + tid], sred[tid]);
}

__global__ void finalize_gate_kernel(const float* __restrict__ gsum,
                                     const float* __restrict__ gfc2_b,
                                     float* __restrict__ g) {
    int i = threadIdx.x;
    if (i < B * E) g[i] = gsum[i] * (1.f / (float)S) + gfc2_b[i & (E - 1)];
}

// ---------------- weight merging: Wm[b] = W + sum_e g[b,e]*tv[e], stored f16 ----------------
__global__ void merge_w_kernel(const float* __restrict__ base,
                               const float* __restrict__ tv,      // [E][n]
                               const float* __restrict__ g,       // [B][E]
                               _Float16* __restrict__ out,        // [B][n]
                               int n) {
    int i = blockIdx.x * blockDim.x + threadIdx.x;
    if (i >= n) return;
    float bs = base[i];
    float t[E];
#pragma unroll
    for (int e = 0; e < E; ++e) t[e] = tv[(size_t)e * n + i];
#pragma unroll
    for (int b = 0; b < B; ++b) {
        float a = bs;
#pragma unroll
        for (int e = 0; e < E; ++e) a += g[b * E + e] * t[e];
        out[(size_t)b * n + i] = (_Float16)a;
    }
}

__global__ void merge_bias_kernel(const float* __restrict__ base,
                                  const float* __restrict__ tv,
                                  const float* __restrict__ g,
                                  float* __restrict__ out, int n) {
    int i = blockIdx.x * blockDim.x + threadIdx.x;
    if (i >= n) return;
    float bs = base[i];
    float t[E];
#pragma unroll
    for (int e = 0; e < E; ++e) t[e] = tv[(size_t)e * n + i];
#pragma unroll
    for (int b = 0; b < B; ++b) {
        float a = bs;
#pragma unroll
        for (int e = 0; e < E; ++e) a += g[b * E + e] * t[e];
        out[(size_t)b * n + i] = a;
    }
}

__global__ void merge_ln_kernel(const float* __restrict__ ln_g, const float* __restrict__ ln_b,
                                const float* __restrict__ tvg, const float* __restrict__ tvb,
                                float* __restrict__ lng, float* __restrict__ lnb) {
    int d = blockIdx.x * blockDim.x + threadIdx.x;
    if (d >= D) return;
    float bg = ln_g[d], bb = ln_b[d];
    float sgacc = 0.f, sbacc = 0.f;
#pragma unroll
    for (int e = 0; e < E; ++e) {
        sgacc += tvg[e * D + d] - bg;
        sbacc += tvb[e * D + d] - bb;
    }
    lng[d] = bg + LN_MERGE * sgacc;
    lnb[d] = bb + LN_MERGE * sbacc;
}

// ---------------- fused FFN: h=relu(x@W1m^T+b1m) in LDS; out=h@W2m^T+b2m; +res; LayerNorm ----------------
__global__ void __launch_bounds__(256, 1)
ffn_kernel(const float* __restrict__ x,            // [B][S][D] fp32 (residual)
           const _Float16* __restrict__ xh,        // [B][S][D] f16
           const _Float16* __restrict__ w1m,       // [B][F][D] f16
           const _Float16* __restrict__ w2m,       // [B][D][F] f16
           const float* __restrict__ b1m,          // [B][F]
           const float* __restrict__ b2m,          // [B][D]
           const float* __restrict__ lng,
           const float* __restrict__ lnb,
           float* __restrict__ out) {
    extern __shared__ __align__(16) char smem[];
    _Float16* xs = (_Float16*)smem;                                  // ST*LDX f16
    _Float16* hs = xs + ST * LDX;                                    // ST*LDH f16
    float* outs  = (float*)(smem + (size_t)(ST * LDX + ST * LDH) * sizeof(_Float16)); // ST*LDO f32

    const int tid  = threadIdx.x;
    const int lane = tid & 31;
    const int wid  = tid >> 5;
    const int b    = blockIdx.x >> 7;
    const int st   = blockIdx.x & 127;
    const int s0   = st * ST;
    const int nl   = lane & 15;
    const int hsel = (lane >> 4) & 1;

    const size_t xbase = ((size_t)b * S + s0) * D;

    // ---- stage f16 x tile: TDM async DMA if available (wave 0 issues), else manual ----
#if __has_builtin(__builtin_amdgcn_tensor_load_to_lds)
    if (wid == 0) {
        tdm_load_x_tile(xh + xbase, __builtin_amdgcn_groupstaticsize());
    }
#else
    for (int c = tid; c < ST * D / 8; c += 256) {
        int row = c >> 6;
        int col = (c & 63) * 8;
        *(v8h*)(xs + row * LDX + col) = *(const v8h*)(xh + xbase + row * D + col);
    }
#endif
    __syncthreads();

    // ---- GEMM1: h[ST, F] = relu(x @ W1m^T + b1m) -> LDS (f16) ----
    // 256 tiles / 8 waves = 32 per wave, processed as 4 batches of 8 N-tiles, k-outer.
    const _Float16* w1b = w1m + (size_t)b * F * D;
    const int m0 = (wid & 1) * 16;
    for (int bo = 0; bo < 4; ++bo) {
        const int nbase = (wid >> 1) * 32 + bo * 8;     // n-tile index base
        v8f acc[8];
#pragma unroll
        for (int j = 0; j < 8; ++j) acc[j] = (v8f){};

        gemm8_pipelined(xs, LDX, m0, w1b, D, nbase * 16, 16, D / 32, lane, acc);

#pragma unroll
        for (int j = 0; j < 8; ++j) {
            const int n0 = (nbase + j) * 16;
            const float bias = b1m[(size_t)b * F + n0 + nl];
#pragma unroll
            for (int r = 0; r < 8; ++r) {
                float v = fmaxf(acc[j][r] + bias, 0.f);
                hs[(m0 + r + hsel * 8) * LDH + (n0 + nl)] = (_Float16)v;
            }
        }
    }
    __syncthreads();

    // ---- GEMM2: out[ST, D] = h @ W2m^T + b2m + x -> LDS (f32); k-outer over 8 N-tiles ----
    const _Float16* w2b = w2m + (size_t)b * D * F;
    {
        v8f acc[8];
#pragma unroll
        for (int j = 0; j < 8; ++j) acc[j] = (v8f){};

        gemm8_pipelined(hs, LDH, m0, w2b, F, (wid >> 1) * 16, 64, F / 32, lane, acc);

#pragma unroll
        for (int j = 0; j < 8; ++j) {
            const int n0   = ((wid >> 1) + 4 * j) * 16;
            const int dcol = n0 + nl;
            const float bias2 = b2m[b * D + dcol];
#pragma unroll
            for (int r = 0; r < 8; ++r) {
                const int row = m0 + r + hsel * 8;
                float res = x[xbase + (size_t)row * D + dcol] + acc[j][r] + bias2;
                outs[row * LDO + dcol] = res;
            }
        }
    }
    __syncthreads();

    // ---- LayerNorm per row (wave32 handles 4 rows) ----
#pragma unroll
    for (int i = 0; i < 4; ++i) {
        const int r = wid * 4 + i;
        float vals[16];
        float sum = 0.f, ss = 0.f;
#pragma unroll
        for (int j = 0; j < 16; ++j) {
            int cc = lane + j * 32;
            float v = outs[r * LDO + cc];
            vals[j] = v;
            sum += v;
            ss  += v * v;
        }
#pragma unroll
        for (int off = 16; off >= 1; off >>= 1) {
            sum += __shfl_xor(sum, off, 32);
            ss  += __shfl_xor(ss,  off, 32);
        }
        const float mu  = sum * (1.f / (float)D);
        const float var = ss * (1.f / (float)D) - mu * mu;
        const float rs  = rsqrtf(var + LN_EPS);
        const size_t ob = ((size_t)b * S + s0 + r) * D;
#pragma unroll
        for (int j = 0; j < 16; ++j) {
            int cc = lane + j * 32;
            out[ob + cc] = (vals[j] - mu) * rs * lng[cc] + lnb[cc];
        }
    }
}

// ---------------- host-side launch ----------------
extern "C" void kernel_launch(void* const* d_in, const int* in_sizes, int n_in,
                              void* d_out, int out_size, void* d_ws, size_t ws_size,
                              hipStream_t stream) {
    const float* x       = (const float*)d_in[0];
    const float* gfc1_w  = (const float*)d_in[1];
    const float* gfc1_b  = (const float*)d_in[2];
    const float* gfc2_w  = (const float*)d_in[3];
    const float* gfc2_b  = (const float*)d_in[4];
    const float* W1      = (const float*)d_in[5];
    const float* b1      = (const float*)d_in[6];
    const float* W2      = (const float*)d_in[7];
    const float* b2      = (const float*)d_in[8];
    const float* ln_g    = (const float*)d_in[9];
    const float* ln_b    = (const float*)d_in[10];
    const float* tv_W1   = (const float*)d_in[11];
    const float* tv_b1   = (const float*)d_in[12];
    const float* tv_W2   = (const float*)d_in[13];
    const float* tv_b2   = (const float*)d_in[14];
    const float* tv_ln_g = (const float*)d_in[15];
    const float* tv_ln_b = (const float*)d_in[16];
    float* out = (float*)d_out;

    // workspace partition (256B aligned slabs)
    char* ws = (char*)d_ws;
    size_t off = 0;
    auto slab = [&](size_t bytes) { size_t r = off; off = (off + bytes + 255) & ~(size_t)255; return r; };
    _Float16* xh    = (_Float16*)(ws + slab((size_t)B * S * D * sizeof(_Float16)));
    _Float16* w1g_h = (_Float16*)(ws + slab((size_t)D * D * sizeof(_Float16)));
    _Float16* w1m_h = (_Float16*)(ws + slab((size_t)B * F * D * sizeof(_Float16)));
    _Float16* w2m_h = (_Float16*)(ws + slab((size_t)B * D * F * sizeof(_Float16)));
    float* gsum = (float*)(ws + slab(B * E * sizeof(float)));
    float* g    = (float*)(ws + slab(B * E * sizeof(float)));
    float* b1m  = (float*)(ws + slab((size_t)B * F * sizeof(float)));
    float* b2m  = (float*)(ws + slab((size_t)B * D * sizeof(float)));
    float* lng  = (float*)(ws + slab(D * sizeof(float)));
    float* lnb  = (float*)(ws + slab(D * sizeof(float)));

    const dim3 blk(256);

    // 1) fp32 -> f16 conversions
    {
        int n4 = B * S * D / 4;
        cvt_f32_to_f16_kernel<<<dim3((n4 + 255) / 256), blk, 0, stream>>>(x, xh, n4);
        int w4 = D * D / 4;
        cvt_f32_to_f16_kernel<<<dim3((w4 + 255) / 256), blk, 0, stream>>>(gfc1_w, w1g_h, w4);
    }

    // 2) gate reduction
    zero_f32_kernel<<<dim3(1), dim3(64), 0, stream>>>(gsum, B * E);
    gate_kernel<<<dim3(B * SBLK), blk, 0, stream>>>(xh, w1g_h, gfc1_b, gfc2_w, gsum);
    finalize_gate_kernel<<<dim3(1), dim3(64), 0, stream>>>(gsum, gfc2_b, g);

    // 3) per-sample weight merges (f16 outputs for WMMA)
    merge_w_kernel<<<dim3(F * D / 256), blk, 0, stream>>>(W1, tv_W1, g, w1m_h, F * D);
    merge_w_kernel<<<dim3(D * F / 256), blk, 0, stream>>>(W2, tv_W2, g, w2m_h, D * F);
    merge_bias_kernel<<<dim3(F / 256), blk, 0, stream>>>(b1, tv_b1, g, b1m, F);
    merge_bias_kernel<<<dim3((D + 255) / 256), blk, 0, stream>>>(b2, tv_b2, g, b2m, D);
    merge_ln_kernel<<<dim3((D + 255) / 256), blk, 0, stream>>>(ln_g, ln_b, tv_ln_g, tv_ln_b, lng, lnb);

    // 4) fused FFN + residual + LayerNorm (231 KB dynamic LDS on CDNA5's 320 KB WGP LDS)
    constexpr size_t SMEM_BYTES =
        (size_t)(ST * LDX + ST * LDH) * sizeof(_Float16) + (size_t)ST * LDO * sizeof(float);
    hipFuncSetAttribute((const void*)ffn_kernel,
                        hipFuncAttributeMaxDynamicSharedMemorySize, (int)SMEM_BYTES);
    ffn_kernel<<<dim3(B * SBLK), blk, SMEM_BYTES, stream>>>(
        x, xh, w1m_h, w2m_h, b1m, b2m, lng, lnb, out);
}
